// MambaWrapper_20289425506408
// MI455X (gfx1250) — compile-verified, matrix operand order checked by reading, not run on hardware
//
#include <hip/hip_runtime.h>
#include <hip/hip_bf16.h>

typedef __attribute__((ext_vector_type(16))) __bf16        bf16x16;
typedef __attribute__((ext_vector_type(8)))  __bf16        bf16x8;
typedef __attribute__((ext_vector_type(8)))  float         floatx8;
typedef __attribute__((ext_vector_type(4)))  unsigned int  u32x4;
typedef __attribute__((ext_vector_type(8)))  int           i32x8;
typedef __attribute__((ext_vector_type(4)))  int           i32x4;

#define MB_B    2
#define MB_L    1024
#define MB_DM   1024
#define MB_DI   2048
#define MB_DTR  64
#define MB_DS   16
#define MB_ROWS (MB_B * MB_L)        /* 2048 tokens */
#define MB_NXZ  (2 * MB_DI)          /* 4096 */
#define MB_NDBL (MB_DTR + 2 * MB_DS) /* 96 */

#if defined(__has_builtin)
#if __has_builtin(__builtin_amdgcn_tensor_load_to_lds) && \
    __has_builtin(__builtin_amdgcn_s_wait_tensorcnt)
#define HAVE_TDM 1
#endif
#endif
#ifndef HAVE_TDM
#define HAVE_TDM 0
#endif

// ---------------------------------------------------------------------------
// elementwise fp32 -> bf16
// ---------------------------------------------------------------------------
__global__ void f32_to_bf16_kernel(const float* __restrict__ src,
                                   __bf16* __restrict__ dst, int n) {
  int i = blockIdx.x * blockDim.x + threadIdx.x;
  if (i < n) dst[i] = (__bf16)src[i];
}

// ---------------------------------------------------------------------------
// LayerNorm over D_MODEL, one row per block, bf16 output for WMMA A operand
// ---------------------------------------------------------------------------
__global__ void layernorm_bf16_kernel(const float* __restrict__ x,
                                      const float* __restrict__ g,
                                      const float* __restrict__ b,
                                      __bf16* __restrict__ out, int D) {
  const int row = blockIdx.x;
  const float* xr = x + (size_t)row * D;
  float s = 0.f, ss = 0.f;
  for (int c = threadIdx.x; c < D; c += blockDim.x) {
    float v = xr[c]; s += v; ss += v * v;
  }
  __shared__ float red0[8];
  __shared__ float red1[8];
  const int lane = threadIdx.x & 31, wv = threadIdx.x >> 5;
  #pragma unroll
  for (int o = 16; o > 0; o >>= 1) { s += __shfl_down(s, o, 32); ss += __shfl_down(ss, o, 32); }
  if (lane == 0) { red0[wv] = s; red1[wv] = ss; }
  __syncthreads();
  if (wv == 0) {
    s  = (lane < (int)(blockDim.x >> 5)) ? red0[lane] : 0.f;
    ss = (lane < (int)(blockDim.x >> 5)) ? red1[lane] : 0.f;
    #pragma unroll
    for (int o = 4; o > 0; o >>= 1) { s += __shfl_down(s, o, 32); ss += __shfl_down(ss, o, 32); }
    if (lane == 0) { red0[0] = s; red1[0] = ss; }
  }
  __syncthreads();
  const float mean = red0[0] / (float)D;
  const float var  = red1[0] / (float)D - mean * mean;
  const float rstd = rsqrtf(var + 1e-5f);
  for (int c = threadIdx.x; c < D; c += blockDim.x) {
    float v = (xr[c] - mean) * rstd * g[c] + b[c];
    out[(size_t)row * D + c] = (__bf16)v;
  }
}

// ---------------------------------------------------------------------------
// WMMA fragment gather per ISA 7.12.2 (16-bit A/B): lanes 0-15 take
// K[0..7]+[16..23], lanes 16-31 take K[8..15]+[24..31] -> two b128 loads.
// Works on both global and LDS pointers.
// ---------------------------------------------------------------------------
__device__ __forceinline__ bf16x16 ld_frag(const __bf16* p) {
  bf16x8 lo = *(const bf16x8*)(p);
  bf16x8 hi = *(const bf16x8*)(p + 16);
  bf16x16 r;
  #pragma unroll
  for (int i = 0; i < 8; ++i) { r[i] = lo[i]; r[i + 8] = hi[i]; }
  return r;
}

#if HAVE_TDM
// ---------------------------------------------------------------------------
// Tensor Data Mover: stage a 2D bf16 tile (tile_d1 rows x tile_d0 elems) of a
// row-major tensor (dim0 = contiguous K, dim0_stride = K) into LDS.
// D# layout per CDNA5 ISA (group0 + group1, 2D case).
// ---------------------------------------------------------------------------
__device__ __forceinline__ void tdm_load_tile_2d(unsigned lds_off,
                                                 const void* gaddr,
                                                 unsigned tensor_d0,
                                                 unsigned tensor_d1,
                                                 unsigned tile_d0,
                                                 unsigned tile_d1,
                                                 unsigned long long stride0) {
  unsigned long long ga = (unsigned long long)(size_t)gaddr;
  u32x4 g0;
  g0[0] = 1u;                                        // count=1, user mode
  g0[1] = lds_off;                                   // lds_addr (bytes)
  g0[2] = (unsigned)(ga & 0xFFFFFFFFu);              // global_addr[31:0]
  g0[3] = (unsigned)((ga >> 32) & 0x01FFFFFFu) | (2u << 30);  // [56:32] | type=2
  i32x8 g1;
  g1[0] = (int)(1u << 16);                           // data_size=1 (2 bytes)
  g1[1] = (int)((tensor_d0 & 0xFFFFu) << 16);        // tensor_dim0[15:0]
  g1[2] = (int)((tensor_d0 >> 16) | ((tensor_d1 & 0xFFFFu) << 16));
  g1[3] = (int)((tensor_d1 >> 16) | (tile_d0 << 16));// tensor_dim1 hi | tile_dim0
  g1[4] = (int)tile_d1;                              // tile_dim1 (tile_dim2=0)
  g1[5] = (int)(unsigned)(stride0 & 0xFFFFFFFFu);    // dim0_stride[31:0]
  g1[6] = (int)(unsigned)((stride0 >> 32) & 0xFFFFu);// dim0_stride[47:32]
  g1[7] = 0;
  i32x4 z4 = {0, 0, 0, 0};
#if __clang_major__ >= 23
  i32x8 z8 = {0, 0, 0, 0, 0, 0, 0, 0};
  __builtin_amdgcn_tensor_load_to_lds(g0, g1, z4, z4, z8, 0);
#else
  __builtin_amdgcn_tensor_load_to_lds(g0, g1, z4, z4, 0);
#endif
}

__device__ __forceinline__ unsigned lds_byte_addr(const void* p) {
  // generic LDS pointer = {aperture[63:32], offset[31:0]}; truncate to offset
  return (unsigned)(size_t)p;
}
#endif  // HAVE_TDM

// ---------------------------------------------------------------------------
// LDS-staged double-buffered bf16 WMMA GEMM: C[M,N] = A[M,K] * W[N,K]^T
// Block: 256 threads (8 waves), tile 128(M) x 128(N), k-step 32.
// Wave w computes rows [w*16, w*16+16) x all 8 n-tiles: 8 WMMA per k-step,
// B fragments preloaded in groups of 4 so WMMAs issue back-to-back after a
// single dscnt wait. TDM (wave 0) double-buffers A/B tiles with
// s_wait_tensorcnt 2 overlapping the next tile's DMA with current compute.
//   EPI 0: plain store   EPI 2: + residual
// ---------------------------------------------------------------------------
template <int EPI>
__global__ __launch_bounds__(256)
void gemm_wmma_lds_kernel(const __bf16* __restrict__ A,
                          const __bf16* __restrict__ W,
                          float* __restrict__ C,
                          const float* __restrict__ residual,
                          int M, int N, int K) {
  __shared__ alignas(16) __bf16 ldsA[2][128][32];
  __shared__ alignas(16) __bf16 ldsB[2][128][32];

  const int nBlocksN = N >> 7;
  const int mb = (int)blockIdx.x / nBlocksN;
  const int nb = (int)blockIdx.x % nBlocksN;
  const int m_base = mb << 7;
  const int n_base = nb << 7;
  const int wave = threadIdx.x >> 5;
  const int lane = threadIdx.x & 31;
  const int half = lane >> 4;
  const int lidx = lane & 15;
  const int ksel = half << 3;

  floatx8 acc[8] = {};

#if HAVE_TDM
  if (wave == 0) {
    tdm_load_tile_2d(lds_byte_addr(&ldsA[0][0][0]),
                     A + (size_t)m_base * K, (unsigned)K, (unsigned)M,
                     32u, 128u, (unsigned long long)K);
    tdm_load_tile_2d(lds_byte_addr(&ldsB[0][0][0]),
                     W + (size_t)n_base * K, (unsigned)K, (unsigned)N,
                     32u, 128u, (unsigned long long)K);
  }
#endif

  int buf = 0;
  for (int k0 = 0; k0 < K; k0 += 32, buf ^= 1) {
#if HAVE_TDM
    __syncthreads();  // all waves done reading buf^1 (previous compute)
    if (wave == 0) {
      if (k0 + 32 < K) {
        tdm_load_tile_2d(lds_byte_addr(&ldsA[buf ^ 1][0][0]),
                         A + (size_t)m_base * K + (k0 + 32), (unsigned)K,
                         (unsigned)M, 32u, 128u, (unsigned long long)K);
        tdm_load_tile_2d(lds_byte_addr(&ldsB[buf ^ 1][0][0]),
                         W + (size_t)n_base * K + (k0 + 32), (unsigned)K,
                         (unsigned)N, 32u, 128u, (unsigned long long)K);
        __builtin_amdgcn_s_wait_tensorcnt((short)2);  // current tile landed
      } else {
        __builtin_amdgcn_s_wait_tensorcnt((short)0);
      }
    }
    __syncthreads();  // publish current buffer
#else
    // cooperative global->LDS staging fallback
    __syncthreads();
    {
      const int tid = threadIdx.x;
      #pragma unroll
      for (int c = tid; c < 512; c += 256) {  // A: 128 rows x 4 b128 chunks
        int row = c >> 2, col = (c & 3) << 3;
        *(bf16x8*)&ldsA[buf][row][col] =
            *(const bf16x8*)(A + (size_t)(m_base + row) * K + k0 + col);
      }
      #pragma unroll
      for (int c = tid; c < 512; c += 256) {  // B: 128 rows x 4 chunks
        int row = c >> 2, col = (c & 3) << 3;
        *(bf16x8*)&ldsB[buf][row][col] =
            *(const bf16x8*)(W + (size_t)(n_base + row) * K + k0 + col);
      }
    }
    __syncthreads();
#endif

    bf16x16 af = ld_frag(&ldsA[buf][wave * 16 + lidx][ksel]);
    #pragma unroll
    for (int g = 0; g < 2; ++g) {
      bf16x16 bfr[4];
      #pragma unroll
      for (int t = 0; t < 4; ++t)
        bfr[t] = ld_frag(&ldsB[buf][(g * 4 + t) * 16 + lidx][ksel]);
      #pragma unroll
      for (int t = 0; t < 4; ++t)
        acc[g * 4 + t] = __builtin_amdgcn_wmma_f32_16x16x32_bf16(
            false, af, false, bfr[t], (short)0, acc[g * 4 + t], false, false);
    }
  }

  const int mbase = m_base + wave * 16 + (half << 3);
  #pragma unroll
  for (int t = 0; t < 8; ++t) {
    const int n = n_base + t * 16 + lidx;
    #pragma unroll
    for (int r = 0; r < 8; ++r) {
      size_t o = (size_t)(mbase + r) * N + n;
      float v = acc[t][r];
      if (EPI == 2) v += residual[o];
      C[o] = v;
    }
  }
}

// ---------------------------------------------------------------------------
// direct-from-global WMMA GEMM for the small projections (N=96 / K=64)
//   EPI 0: plain   EPI 1: softplus(v + bias[n])
// ---------------------------------------------------------------------------
template <int NTILES, int EPI>
__global__ void gemm_wmma_bf16_kernel(const __bf16* __restrict__ A,
                                      const __bf16* __restrict__ W,
                                      float* __restrict__ C,
                                      const float* __restrict__ bias,
                                      int M, int N, int K) {
  const int lane          = threadIdx.x & 31;
  const int wavesPerBlock = blockDim.x >> 5;
  const int tileId        = blockIdx.x * wavesPerBlock + (threadIdx.x >> 5);
  const int ntilesN       = N / (16 * NTILES);
  const int totalTiles    = (M >> 4) * ntilesN;
  if (tileId >= totalTiles) return;

  const int m0   = (tileId / ntilesN) << 4;
  const int n0   = (tileId % ntilesN) * (16 * NTILES);
  const int half = lane >> 4;
  const int lidx = lane & 15;
  const int ksel = half << 3;

  const __bf16* pa = A + (size_t)(m0 + lidx) * K + ksel;
  const __bf16* pw[NTILES];
  #pragma unroll
  for (int t = 0; t < NTILES; ++t)
    pw[t] = W + (size_t)(n0 + t * 16 + lidx) * K + ksel;

  floatx8 acc[NTILES] = {};

  for (int k0 = 0; k0 < K; k0 += 32) {
    __builtin_prefetch(pa + k0 + 512, 0, 1);  // global_prefetch_b8
    __builtin_prefetch(pw[0] + k0 + 512, 0, 1);
    bf16x16 af = ld_frag(pa + k0);
    bf16x16 bfr[NTILES];
    #pragma unroll
    for (int t = 0; t < NTILES; ++t) bfr[t] = ld_frag(pw[t] + k0);
    #pragma unroll
    for (int t = 0; t < NTILES; ++t)
      acc[t] = __builtin_amdgcn_wmma_f32_16x16x32_bf16(
          false, af, false, bfr[t], (short)0, acc[t], false, false);
  }

  const int mbase = m0 + (half << 3);
  #pragma unroll
  for (int t = 0; t < NTILES; ++t) {
    const int n = n0 + t * 16 + lidx;
    #pragma unroll
    for (int r = 0; r < 8; ++r) {
      size_t o = (size_t)(mbase + r) * N + n;
      float v = acc[t][r];
      if (EPI == 1) { float xv = v + bias[n]; v = (xv > 20.f) ? xv : log1pf(__expf(xv)); }
      C[o] = v;
    }
  }
}

// ---------------------------------------------------------------------------
// causal depthwise conv (d_conv=4) + bias + SiLU; reads u-half of xz
// ---------------------------------------------------------------------------
__global__ void conv_silu_kernel(const float* __restrict__ xz,
                                 const float* __restrict__ cw,
                                 const float* __restrict__ cb,
                                 float* __restrict__ uf,
                                 __bf16* __restrict__ ub) {
  int i = blockIdx.x * blockDim.x + threadIdx.x;
  if (i >= MB_B * MB_L * MB_DI) return;
  const int d = i % MB_DI;
  const int l = (i / MB_DI) % MB_L;
  const int b = i / (MB_DI * MB_L);
  float acc = cb[d];
  #pragma unroll
  for (int j = 0; j < 4; ++j) {
    int ll = l + j - 3;
    if (ll >= 0)
      acc += cw[d * 4 + j] * xz[(size_t)(b * MB_L + ll) * MB_NXZ + d];
  }
  const float s = acc / (1.f + __expf(-acc));
  const size_t o = (size_t)(b * MB_L + l) * MB_DI + d;
  uf[o] = s;
  ub[o] = (__bf16)s;
}

// dt = x_dbl[:, 0:64] -> contiguous bf16 for the dt_proj WMMA GEMM
__global__ void extract_dt_kernel(const float* __restrict__ xdbl,
                                  __bf16* __restrict__ dt, int rows) {
  int i = blockIdx.x * blockDim.x + threadIdx.x;
  if (i >= rows * MB_DTR) return;
  const int r = i / MB_DTR, c = i % MB_DTR;
  dt[i] = (__bf16)xdbl[(size_t)r * MB_NDBL + c];
}

// ---------------------------------------------------------------------------
// selective scan: one thread per (b, d) channel, h[16] in VGPRs, fp32
// ---------------------------------------------------------------------------
__global__ void selective_scan_kernel(const float* __restrict__ delta,
                                      const float* __restrict__ uf,
                                      const float* __restrict__ xdbl,
                                      const float* __restrict__ A_log,
                                      const float* __restrict__ Dskip,
                                      float* __restrict__ y) {
  int tid = blockIdx.x * blockDim.x + threadIdx.x;
  if (tid >= MB_B * MB_DI) return;
  const int b = tid / MB_DI, d = tid % MB_DI;
  float A[MB_DS], h[MB_DS];
  #pragma unroll
  for (int n = 0; n < MB_DS; ++n) { A[n] = -__expf(A_log[d * MB_DS + n]); h[n] = 0.f; }
  const float Dd = Dskip[d];
  for (int l = 0; l < MB_L; ++l) {
    const size_t row = (size_t)b * MB_L + l;
    const float dv = delta[row * MB_DI + d];
    const float uv = uf[row * MB_DI + d];
    const float* Bt = xdbl + row * MB_NDBL + MB_DTR;
    const float* Ct = Bt + MB_DS;
    const float du = dv * uv;
    float acc = 0.f;
    #pragma unroll
    for (int n = 0; n < MB_DS; ++n) {
      float dA = __expf(dv * A[n]);
      h[n] = dA * h[n] + du * Bt[n];
      acc += h[n] * Ct[n];
    }
    y[row * MB_DI + d] = acc + uv * Dd;
  }
}

// y_gated = (scan_y) * silu(z)  -> bf16 for out_proj GEMM
__global__ void gate_kernel(const float* __restrict__ y,
                            const float* __restrict__ xz,
                            __bf16* __restrict__ yg) {
  int i = blockIdx.x * blockDim.x + threadIdx.x;
  if (i >= MB_ROWS * MB_DI) return;
  const int r = i / MB_DI, d = i % MB_DI;
  const float z = xz[(size_t)r * MB_NXZ + MB_DI + d];
  const float g = z / (1.f + __expf(-z));
  yg[i] = (__bf16)(y[i] * g);
}

// ---------------------------------------------------------------------------
extern "C" void kernel_launch(void* const* d_in, const int* in_sizes, int n_in,
                              void* d_out, int out_size, void* d_ws, size_t ws_size,
                              hipStream_t stream) {
  const float* x      = (const float*)d_in[0];
  const float* ln_g   = (const float*)d_in[1];
  const float* ln_b   = (const float*)d_in[2];
  const float* w_in   = (const float*)d_in[3];
  const float* conv_w = (const float*)d_in[4];
  const float* conv_b = (const float*)d_in[5];
  const float* w_xp   = (const float*)d_in[6];
  const float* w_dt   = (const float*)d_in[7];
  const float* dt_b   = (const float*)d_in[8];
  const float* A_log  = (const float*)d_in[9];
  const float* Dskip  = (const float*)d_in[10];
  const float* w_out  = (const float*)d_in[11];
  float* out = (float*)d_out;

  char* ws = (char*)d_ws;
  size_t off = 0;
  auto carve = [&](size_t bytes) -> void* {
    void* p = ws + off;
    off += (bytes + 255) & ~(size_t)255;
    return p;
  };

  __bf16* xn_b   = (__bf16*)carve((size_t)MB_ROWS * MB_DM * 2);
  __bf16* win_b  = (__bf16*)carve((size_t)MB_NXZ * MB_DM * 2);
  __bf16* wxp_b  = (__bf16*)carve((size_t)MB_NDBL * MB_DI * 2);
  __bf16* wdt_b  = (__bf16*)carve((size_t)MB_DI * MB_DTR * 2);
  __bf16* wout_b = (__bf16*)carve((size_t)MB_DM * MB_DI * 2);
  float*  xz     = (float*) carve((size_t)MB_ROWS * MB_NXZ * 4);
  float*  uf     = (float*) carve((size_t)MB_ROWS * MB_DI * 4);
  __bf16* ub     = (__bf16*)carve((size_t)MB_ROWS * MB_DI * 2);
  float*  xdbl   = (float*) carve((size_t)MB_ROWS * MB_NDBL * 4);
  __bf16* dt_bf  = (__bf16*)carve((size_t)MB_ROWS * MB_DTR * 2);
  float*  deltaf = (float*) carve((size_t)MB_ROWS * MB_DI * 4);
  float*  yf     = (float*) carve((size_t)MB_ROWS * MB_DI * 4);
  __bf16* yg     = (__bf16*)carve((size_t)MB_ROWS * MB_DI * 2);

  auto cvt = [&](const float* s, __bf16* d, int n) {
    f32_to_bf16_kernel<<<(n + 255) / 256, 256, 0, stream>>>(s, d, n);
  };
  cvt(w_in,  win_b,  MB_NXZ * MB_DM);
  cvt(w_xp,  wxp_b,  MB_NDBL * MB_DI);
  cvt(w_dt,  wdt_b,  MB_DI * MB_DTR);
  cvt(w_out, wout_b, MB_DM * MB_DI);

  layernorm_bf16_kernel<<<MB_ROWS, 256, 0, stream>>>(x, ln_g, ln_b, xn_b, MB_DM);

  {  // xz = xn @ in_proj^T : 2048 x 4096, K=1024  (TDM + LDS + WMMA)
    int blocks = (MB_ROWS / 128) * (MB_NXZ / 128);
    gemm_wmma_lds_kernel<0><<<blocks, 256, 0, stream>>>(
        xn_b, win_b, xz, nullptr, MB_ROWS, MB_NXZ, MB_DM);
  }

  conv_silu_kernel<<<(MB_B * MB_L * MB_DI + 255) / 256, 256, 0, stream>>>(
      xz, conv_w, conv_b, uf, ub);

  {  // x_dbl = u @ x_proj^T : 2048 x 96, K=2048 (direct WMMA)
    int tiles = (MB_ROWS / 16) * (MB_NDBL / 32);
    gemm_wmma_bf16_kernel<2, 0><<<(tiles + 7) / 8, 256, 0, stream>>>(
        ub, wxp_b, xdbl, nullptr, MB_ROWS, MB_NDBL, MB_DI);
  }

  extract_dt_kernel<<<(MB_ROWS * MB_DTR + 255) / 256, 256, 0, stream>>>(
      xdbl, dt_bf, MB_ROWS);

  {  // delta = softplus(dt @ dt_proj^T + b) : 2048 x 2048, K=64 (direct WMMA)
    int tiles = (MB_ROWS / 16) * (MB_DI / 64);
    gemm_wmma_bf16_kernel<4, 1><<<tiles / 8, 256, 0, stream>>>(
        dt_bf, wdt_b, deltaf, dt_b, MB_ROWS, MB_DI, MB_DTR);
  }

  selective_scan_kernel<<<(MB_B * MB_DI + 255) / 256, 256, 0, stream>>>(
      deltaf, uf, xdbl, A_log, Dskip, yf);

  gate_kernel<<<(MB_ROWS * MB_DI + 255) / 256, 256, 0, stream>>>(yf, xz, yg);

  {  // out = y_g @ out_proj^T + residual : 2048 x 1024, K=2048 (TDM + LDS)
    int blocks = (MB_ROWS / 128) * (MB_DM / 128);
    gemm_wmma_lds_kernel<2><<<blocks, 256, 0, stream>>>(
        yg, wout_b, out, x, MB_ROWS, MB_DM, MB_DI);
  }

  (void)in_sizes; (void)n_in; (void)out_size; (void)ws_size;
}
